// ContextAttention_15470472200703
// MI455X (gfx1250) — matrix operand order repaired and checked by reading, not compiled, not run on hardware
//
#include <hip/hip_runtime.h>
#include <hip/hip_bf16.h>
#include <stdint.h>

// ---------------------------------------------------------------------------
// MI455X (gfx1250) fused context-attention:
//   k0: W fp32 -> bf16 (2.3MB -> 1.15MB, L2 resident)
//   kA: scores[h][i] = <tanh(x_i W^T + b) restricted to head h, u_c[h]>
//       bf16 WMMA (v_wmma_f32_16x16x32_bf16), f32 accum, u_t never stored.
//       W chunks streamed to LDS with global_load_async_to_lds_b128.
//   kB: per-head softmax stats (max, 1/sum) over 131072; zeroes d_out
//   kC: m[h][d] = sum_i softmax(scores)[h][i] * x[i][h*128+d]  (exact fp32)
// HBM traffic ~ 2 passes over x (805MB) ~= 35us @ 23.3 TB/s; bf16 WMMA for
// the 154.6 GFLOP GEMM is roughly balanced against that.
// Workspace layout (needs ~4.2MB):
//   [0, 3145728)        scores  f32 [6][131072]
//   [3145728, +48)      stats   f32 mx[6], rsum[6]
//   [3146240, +1179648) W bf16  [768][768]
// ---------------------------------------------------------------------------

#define B_ROWS    131072
#define Z_DIM     768
#define NH        6
#define HS        128
#define ROWS_WG   128
#define KC        32
#define XS_STRIDE 40   // ushorts/row: 32 data + 8 pad -> 80B, 16B aligned
#define WL_STRIDE 40
#define L2E       1.4426950408889634f

typedef __attribute__((ext_vector_type(16))) __bf16       v16bf;
typedef __attribute__((ext_vector_type(8)))  float        v8f;
typedef __attribute__((ext_vector_type(4)))  float        f32x4;
typedef __attribute__((ext_vector_type(4)))  unsigned int u32x4;
typedef __attribute__((ext_vector_type(2)))  unsigned int u32x2;

static __device__ __forceinline__ unsigned f2bf_rne(float f) {
  unsigned u = __float_as_uint(f);
  return (u + 0x7FFFu + ((u >> 16) & 1u)) >> 16;
}
// pack two f32 -> packed bf16 pair (a in [15:0], b in [31:16])
static __device__ __forceinline__ unsigned pack2bf(float a, float b) {
#if __has_builtin(__builtin_amdgcn_cvt_pk_bf16_f32)
  typedef __attribute__((ext_vector_type(2))) __bf16 v2bf;
  union { v2bf v; unsigned u; } p;
  p.v = __builtin_amdgcn_cvt_pk_bf16_f32(a, b);   // v_cvt_pk_bf16_f32
  return p.u;
#else
  return f2bf_rne(a) | (f2bf_rne(b) << 16);
#endif
}
// tanh(x) = 1 - 2/(e^(2x)+1); v_exp + v_rcp, no IEEE div expansion
static __device__ __forceinline__ float fast_tanh(float x) {
  float cx = fminf(fmaxf(x, -15.f), 15.f);
  float e  = __builtin_amdgcn_exp2f(cx * (2.0f * L2E));   // e^(2x)
  float r  = __builtin_amdgcn_rcpf(e + 1.0f);
  return __builtin_fmaf(-2.0f, r, 1.0f);
}
static __device__ __forceinline__ float swz_xor(float v, int imm_unused);
// one level of a multi-value butterfly merge: lanes with (lane&M)==0 end up
// owning sum-group of `a`, lanes with (lane&M)!=0 own group of `b`.
template <int M>
static __device__ __forceinline__ float merge2(float a, float b, int lane) {
  bool hi    = (lane & M) != 0;
  float keep = hi ? b : a;
  float send = hi ? a : b;
  float got  = __int_as_float(
      __builtin_amdgcn_ds_swizzle(__float_as_int(send), (M << 10) | 0x1F));
  return keep + got;
}
static __device__ __forceinline__ float xor16_sum(float v) {
  v += __int_as_float(__builtin_amdgcn_ds_swizzle(__float_as_int(v), 0x041F)); // xor 1
  v += __int_as_float(__builtin_amdgcn_ds_swizzle(__float_as_int(v), 0x081F)); // xor 2
  v += __int_as_float(__builtin_amdgcn_ds_swizzle(__float_as_int(v), 0x101F)); // xor 4
  v += __int_as_float(__builtin_amdgcn_ds_swizzle(__float_as_int(v), 0x201F)); // xor 8
  return v;
}

// ---------------- k0: W fp32 -> bf16 -------------------------------------
__global__ __launch_bounds__(256) void wcvt_kernel(const float* __restrict__ W,
                                                   unsigned short* __restrict__ wbf) {
  int idx = (blockIdx.x * 256 + threadIdx.x) * 4;
  f32x4 v = *(const f32x4*)(W + idx);
  u32x2 p;
  p[0] = pack2bf(v[0], v[1]);
  p[1] = pack2bf(v[2], v[3]);
  *(u32x2*)(wbf + idx) = p;
}

// ---------------- kA: GEMM(tanh) + per-head score ------------------------
__global__ __launch_bounds__(256) void gemm_score_kernel(
    const float* __restrict__ x, const unsigned short* __restrict__ wbf,
    const float* __restrict__ b, const float* __restrict__ uc,
    float* __restrict__ scores) {
  __shared__ unsigned short xs[ROWS_WG * XS_STRIDE];  // 10.0 KB, x chunk bf16
  __shared__ unsigned short wl[HS * WL_STRIDE];       // 10.0 KB, W chunk bf16
  __shared__ float score_lds[ROWS_WG];

  const int  t     = threadIdx.x;
  const int  lane  = t & 31;
  const int  wv    = t >> 5;
  const int  rquad = wv & 3;   // rows  rquad*32 .. +31
  const int  chalf = wv >> 2;  // cols  chalf*64 .. +63 (within 128-col head)
  const long rowBase = (long)blockIdx.x * ROWS_WG;

  for (int cb = 0; cb < NH; ++cb) {            // column block == head
    __syncthreads();
    if (t < ROWS_WG) score_lds[t] = 0.f;

    v8f acc[2][4];
    #pragma unroll
    for (int rt = 0; rt < 2; ++rt)
      #pragma unroll
      for (int ct = 0; ct < 4; ++ct)
        acc[rt][ct] = (v8f){0.f, 0.f, 0.f, 0.f, 0.f, 0.f, 0.f, 0.f};

    for (int kb = 0; kb < Z_DIM / KC; ++kb) {
      __syncthreads();  // previous compute done before LDS overwrite
      // ---- stage x[128][32] fp32 -> bf16 into LDS (1024 float4 ops) ----
      #pragma unroll
      for (int j = 0; j < 4; ++j) {
        int o = t + j * 256;
        int row = o >> 3, kq = o & 7;
        const float* gp = x + (rowBase + row) * Z_DIM + kb * KC + kq * 4;
        f32x4 v = *(const f32x4*)gp;
        u32x2 p;
        p[0] = pack2bf(v[0], v[1]);
        p[1] = pack2bf(v[2], v[3]);
        *(u32x2*)&xs[row * XS_STRIDE + kq * 4] = p;
      }
      // ---- stage W[128 cols][32 k] bf16 via async copy to LDS ----
      #pragma unroll
      for (int j = 0; j < 2; ++j) {
        int o = t + j * 256;              // 512 x 16B
        int col = o >> 2, q = o & 3;
        const unsigned short* gp =
            wbf + (size_t)(cb * HS + col) * Z_DIM + kb * KC + q * 8;
        unsigned lds_addr = (unsigned)(uintptr_t)&wl[col * WL_STRIDE + q * 8];
        asm volatile("global_load_async_to_lds_b128 %0, %1, off"
                     :: "v"(lds_addr), "v"((unsigned long long)(uintptr_t)gp)
                     : "memory");
      }
      asm volatile("s_wait_asynccnt 0" ::: "memory");
      __syncthreads();

      // ---- preload ALL fragments, then 8 back-to-back WMMAs ----
      v16bf A[2], Bf[4];
      #pragma unroll
      for (int rt = 0; rt < 2; ++rt) {
        int row = rquad * 32 + rt * 16 + (lane & 15);
        union { v16bf v; u32x4 q[2]; } ua;
        ua.q[0] = *(const u32x4*)&xs[row * XS_STRIDE + (lane >> 4) * 8];       // K 0..7 / 8..15
        ua.q[1] = *(const u32x4*)&xs[row * XS_STRIDE + 16 + (lane >> 4) * 8];  // K 16..23 / 24..31
        A[rt] = ua.v;
      }
      #pragma unroll
      for (int ct = 0; ct < 4; ++ct) {
        int col = chalf * 64 + ct * 16 + (lane & 15);
        const u32x4* pb = (const u32x4*)&wl[col * WL_STRIDE + (lane >> 4) * 8];
        union { v16bf v; u32x4 q[2]; } ub;
        ub.q[0] = pb[0];  // K 0..7  / 8..15
        ub.q[1] = pb[2];  // K 16..23 / 24..31
        Bf[ct] = ub.v;
      }
      #pragma unroll
      for (int ct = 0; ct < 4; ++ct)
        #pragma unroll
        for (int rt = 0; rt < 2; ++rt)
          acc[rt][ct] = __builtin_amdgcn_wmma_f32_16x16x32_bf16(
              false, A[rt], false, Bf[ct], (short)0, acc[rt][ct], false, false);
    }

    // ---- epilogue: tanh, fold ct in regs, 8-value merge tree, ds_add ----
    float u4[4], b4[4];
    #pragma unroll
    for (int ct = 0; ct < 4; ++ct) {
      int ncol = chalf * 64 + ct * 16 + (lane & 15);  // 0..127 within head
      u4[ct] = uc[cb * HS + ncol];
      b4[ct] = b[cb * HS + ncol];
    }
    #pragma unroll
    for (int rt = 0; rt < 2; ++rt) {
      float sv[8];
      #pragma unroll
      for (int v = 0; v < 8; ++v) sv[v] = 0.f;
      #pragma unroll
      for (int ct = 0; ct < 4; ++ct)
        #pragma unroll
        for (int v = 0; v < 8; ++v)
          sv[v] = __builtin_fmaf(fast_tanh(acc[rt][ct][v] + b4[ct]), u4[ct], sv[v]);
      // merge tree: after xor 1,2,4 each lane owns row index (lane&7);
      // xor 8 completes the 16-lane sum.
      float t0 = merge2<1>(sv[0], sv[1], lane);
      float t1 = merge2<1>(sv[2], sv[3], lane);
      float t2 = merge2<1>(sv[4], sv[5], lane);
      float t3 = merge2<1>(sv[6], sv[7], lane);
      float u0 = merge2<2>(t0, t1, lane);
      float u1 = merge2<2>(t2, t3, lane);
      float w0 = merge2<4>(u0, u1, lane);
      w0 += __int_as_float(
          __builtin_amdgcn_ds_swizzle(__float_as_int(w0), (8 << 10) | 0x1F));
      if ((lane & 8) == 0)  // 16 active lanes, each owns one distinct row
        atomicAdd(&score_lds[rquad * 32 + rt * 16 + (lane & 7) + ((lane >> 4) << 3)], w0);
    }
    __syncthreads();
    if (t < ROWS_WG)
      scores[(size_t)cb * B_ROWS + rowBase + t] = score_lds[t];
  }
}

// ---------------- kB: softmax stats per head; zero d_out ------------------
__global__ __launch_bounds__(256) void softmax_stats_kernel(
    const float* __restrict__ scores, float* __restrict__ stats,
    float* __restrict__ out) {
  const int h = blockIdx.x, t = threadIdx.x, lane = t & 31, wv = t >> 5;
  __shared__ float red[8];
  const float* s = scores + (size_t)h * B_ROWS;

  float mx = -3.402823466e38f;
  for (int i = t; i < B_ROWS; i += 256) mx = fmaxf(mx, s[i]);
  mx = fmaxf(mx, __int_as_float(__builtin_amdgcn_ds_swizzle(__float_as_int(mx), 0x041F)));
  mx = fmaxf(mx, __int_as_float(__builtin_amdgcn_ds_swizzle(__float_as_int(mx), 0x081F)));
  mx = fmaxf(mx, __int_as_float(__builtin_amdgcn_ds_swizzle(__float_as_int(mx), 0x101F)));
  mx = fmaxf(mx, __int_as_float(__builtin_amdgcn_ds_swizzle(__float_as_int(mx), 0x201F)));
  mx = fmaxf(mx, __int_as_float(__builtin_amdgcn_ds_swizzle(__float_as_int(mx), 0x401F)));
  if (lane == 0) red[wv] = mx;
  __syncthreads();
  mx = red[0];
  #pragma unroll
  for (int w = 1; w < 8; ++w) mx = fmaxf(mx, red[w]);
  __syncthreads();

  float sum = 0.f;
  for (int i = t; i < B_ROWS; i += 256)
    sum += __builtin_amdgcn_exp2f((s[i] - mx) * L2E);
  sum = xor16_sum(sum);
  sum += __int_as_float(__builtin_amdgcn_ds_swizzle(__float_as_int(sum), 0x401F));
  if (lane == 0) red[wv] = sum;
  __syncthreads();
  sum = 0.f;
  #pragma unroll
  for (int w = 0; w < 8; ++w) sum += red[w];

  if (t == 0) { stats[h] = mx; stats[NH + h] = 1.f / sum; }
  if (t < HS) out[h * HS + t] = 0.f;  // zero for kC atomics
}

// ---------------- kC: m = sum_i a[h][i] * x[i, head slice] ----------------
__global__ __launch_bounds__(256) void weighted_sum_kernel(
    const float* __restrict__ x, const float* __restrict__ scores,
    const float* __restrict__ stats, float* __restrict__ out) {
  const int t = threadIdx.x;
  __shared__ float wch[32][8];  // 6 heads used, padded
  const int  ROWS_BLK = B_ROWS / 256;  // 512
  const long rowBase0 = (long)blockIdx.x * ROWS_BLK;

  float mxh = 0.f, rsh = 0.f;
  int rr = 0, hh = 0;
  if (t < 192) {
    rr = t / 6; hh = t - rr * 6;
    mxh = stats[hh]; rsh = stats[NH + hh];
  }

  float acc0 = 0.f, acc1 = 0.f, acc2 = 0.f;
  const int h0 = t >> 7, h1 = (t + 256) >> 7, h2 = (t + 512) >> 7;

  for (int ch = 0; ch < ROWS_BLK / 32; ++ch) {
    long rb = rowBase0 + ch * 32;
    __syncthreads();
    if (t < 192)
      wch[rr][hh] =
          __builtin_amdgcn_exp2f((scores[(size_t)hh * B_ROWS + rb + rr] - mxh) * L2E) * rsh;
    __syncthreads();
    for (int r = 0; r < 32; ++r) {
      const float* xp = x + (rb + r) * Z_DIM;
      acc0 += wch[r][h0] * xp[t];
      acc1 += wch[r][h1] * xp[t + 256];
      acc2 += wch[r][h2] * xp[t + 512];
    }
  }
  atomicAdd(&out[t],       acc0);
  atomicAdd(&out[t + 256], acc1);
  atomicAdd(&out[t + 512], acc2);
}

// ---------------- launcher ------------------------------------------------
extern "C" void kernel_launch(void* const* d_in, const int* in_sizes, int n_in,
                              void* d_out, int out_size, void* d_ws, size_t ws_size,
                              hipStream_t stream) {
  const float* x  = (const float*)d_in[0];
  const float* W  = (const float*)d_in[1];
  const float* b  = (const float*)d_in[2];
  const float* uc = (const float*)d_in[3];
  float* out = (float*)d_out;

  char* ws = (char*)d_ws;
  float*          scores = (float*)ws;                  // 3,145,728 B
  float*          stats  = (float*)(ws + 3145728);      // 48 B
  unsigned short* wbf    = (unsigned short*)(ws + 3146240);  // 1,179,648 B

  wcvt_kernel<<<(Z_DIM * Z_DIM) / 1024, 256, 0, stream>>>(W, wbf);
  gemm_score_kernel<<<B_ROWS / ROWS_WG, 256, 0, stream>>>(x, wbf, b, uc, scores);
  softmax_stats_kernel<<<NH, 256, 0, stream>>>(scores, stats, out);
  weighted_sum_kernel<<<256, 256, 0, stream>>>(x, scores, stats, out);
}
